// UniversalContext_31275951850298
// MI455X (gfx1250) — compile-verified, matrix operand order checked by reading, not run on hardware
//
#include <hip/hip_runtime.h>
#include <hip/hip_bf16.h>

typedef __attribute__((ext_vector_type(16))) _Float16 v16h;
typedef __attribute__((ext_vector_type(8)))  _Float16 h8;
typedef __attribute__((ext_vector_type(8)))  float    v8f;

// Problem constants (from reference setup_inputs)
#define BB 8
#define CC 192
#define HH 64
#define WW 64
#define HWSZ (HH * WW)          // 4096
#define GG 8
#define KK 512
#define DD 24
#define DP 32                   // padded K-dim for WMMA (24 -> 32)
#define LSTRIDE 48              // LDS row stride in halves (96B, 16B-aligned h8 loads)
#define ROWS_PER_BLOCK 128      // 8 waves x 16 query rows
#define NTILES (KK / 16)        // 32 code tiles per group

__global__ __launch_bounds__(256)
void vq_nearest_wmma_kernel(const float* __restrict__ y,
                            const float* __restrict__ codebooks,
                            float* __restrict__ out_uc,
                            float* __restrict__ out_yba,
                            float* __restrict__ out_ci) {
    __shared__ __align__(16) _Float16 cbh[KK * LSTRIDE];             // f16 codebook, K-padded
    __shared__ __align__(16) _Float16 xth[ROWS_PER_BLOCK * LSTRIDE]; // f16 query tile
    __shared__ float e2s[KK];                                        // per-code squared norms
    __shared__ int   bestk[ROWS_PER_BLOCK];                          // per-row argmin

    const int tid  = threadIdx.x;
    const int tile = blockIdx.x % (HWSZ / ROWS_PER_BLOCK);   // 0..31
    const int g    = (blockIdx.x / (HWSZ / ROWS_PER_BLOCK)) % GG;
    const int b    = blockIdx.x / ((HWSZ / ROWS_PER_BLOCK) * GG);
    const int rowBase = tile * ROWS_PER_BLOCK;

    const float* cbg = codebooks + (size_t)g * KK * DD;
    __builtin_prefetch(cbg, 0, 1);   // global_prefetch_b8: warm the group's codebook

    // ---- Stage codebook (f16, K padded to 32) into LDS ----
    for (int s = tid; s < KK * DP; s += 256) {
        const int k = s >> 5;
        const int j = s & 31;
        const float v = (j < DD) ? cbg[k * DD + j] : 0.0f;
        cbh[k * LSTRIDE + j] = (_Float16)v;
    }
    // ---- Per-code squared norms (f32) ----
    for (int k = tid; k < KK; k += 256) {
        float acc = 0.0f;
        #pragma unroll
        for (int j = 0; j < DD; ++j) {
            const float v = cbg[k * DD + j];
            acc += v * v;
        }
        e2s[k] = acc;
    }
    // ---- Stage 128 query rows (f16, padded) into LDS; coalesced in spatial p ----
    for (int s = tid; s < DP * ROWS_PER_BLOCK; s += 256) {
        const int j = s >> 7;        // channel within group (0..31)
        const int r = s & 127;       // local row
        float v = 0.0f;
        if (j < DD) {
            const size_t yo = ((size_t)b * CC + g * DD + j) * HWSZ + rowBase + r;
            v = y[yo];
        }
        xth[r * LSTRIDE + j] = (_Float16)v;
    }
    __syncthreads();

    // ---- WMMA distance search: each wave owns one 16-row M-tile ----
    const int w    = tid >> 5;       // wave id 0..7
    const int lane = tid & 31;
    const int hi   = lane >> 4;      // 0: lanes 0-15, 1: lanes 16-31
    const int rlow = lane & 15;
    const int kbase = hi ? 8 : 0;    // WMMA f16 A/B lane->K mapping

    // A fragment: 16 halves = K {kbase..kbase+7} ++ K {kbase+16..kbase+23}
    const int arow = w * 16 + rlow;
    const h8 alo = *(const h8*)&xth[arow * LSTRIDE + kbase];
    const h8 ahi = *(const h8*)&xth[arow * LSTRIDE + kbase + 16];
    v16h afrag;
    #pragma unroll
    for (int t = 0; t < 8; ++t) { afrag[t] = alo[t]; afrag[t + 8] = ahi[t]; }

    float bd[8];
    int   bi[8];
    #pragma unroll
    for (int r = 0; r < 8; ++r) { bd[r] = 3.4e38f; bi[r] = 0; }

    // Software-pipelined B-fragment / e2 loads: issue tile nt+1's LDS loads
    // before the WMMA of tile nt so ds latency hides under the matrix op.
    const _Float16* cbp = &cbh[rlow * LSTRIDE + kbase];
    h8 blo = *(const h8*)(cbp);
    h8 bhi = *(const h8*)(cbp + 16);
    float e2n = e2s[rlow];

    for (int nt = 0; nt < NTILES; ++nt) {
        v16h bfrag;
        #pragma unroll
        for (int t = 0; t < 8; ++t) { bfrag[t] = blo[t]; bfrag[t + 8] = bhi[t]; }
        const float e2cur = e2n;
        const int   n     = nt * 16 + rlow;   // code index this lane scores

        if (nt + 1 < NTILES) {                // prefetch next tile's operands
            cbp += 16 * LSTRIDE;
            blo = *(const h8*)(cbp);
            bhi = *(const h8*)(cbp + 16);
            e2n = e2s[n + 16];
        }

        v8f czero = {};
        const v8f c = __builtin_amdgcn_wmma_f32_16x16x32_f16(
            false, afrag, false, bfrag, (short)0, czero, false, false);

        #pragma unroll
        for (int r = 0; r < 8; ++r) {
            // dist(M=r+8*hi, N=n) up to the constant ||x||^2 (irrelevant for argmin)
            const float dist = e2cur - 2.0f * c[r];
            const bool take = dist < bd[r];    // strict <: keeps first (smallest n)
            bd[r] = take ? dist : bd[r];
            bi[r] = take ? n    : bi[r];
        }
    }

    // ---- Cross-lane argmin over the 16 lanes sharing each row (tie -> smaller idx) ----
    #pragma unroll
    for (int r = 0; r < 8; ++r) {
        float d = bd[r];
        int   i = bi[r];
        #pragma unroll
        for (int m = 1; m <= 8; m <<= 1) {   // masks 1,2,4,8 stay within each 16-lane half
            const float od = __shfl_xor(d, m, 32);
            const int   oi = __shfl_xor(i, m, 32);
            // branchless: avoid exec-mask divergence from short-circuit ||
            const bool take = (od < d) | ((od == d) & (oi < i));
            d = take ? od : d;
            i = take ? oi : i;
        }
        bd[r] = d; bi[r] = i;
    }
    if (rlow == 0) {
        #pragma unroll
        for (int r = 0; r < 8; ++r) {
            bestk[w * 16 + r + (hi ? 8 : 0)] = bi[r];
        }
    }
    __syncthreads();

    // ---- Emit outputs (block-cooperative, coalesced in p) ----
    for (int s = tid; s < ROWS_PER_BLOCK * DD; s += 256) {
        const int j = s / ROWS_PER_BLOCK;     // channel within group
        const int r = s % ROWS_PER_BLOCK;     // local row
        const int p = rowBase + r;
        const int ks = bestk[r];
        const float q  = cbg[ks * DD + j];    // full-precision codebook value
        const size_t yo = ((size_t)b * CC + g * DD + j) * HWSZ + p;
        const float yv = y[yo];
        out_yba[yo] = q;
        out_uc[yo]  = yv + (q - yv);          // match reference rounding of STE forward
    }
    if (tid < ROWS_PER_BLOCK) {
        const int p = rowBase + tid;
        out_ci[((size_t)b * GG + g) * HWSZ + p] = (float)bestk[tid];
    }
}

extern "C" void kernel_launch(void* const* d_in, const int* in_sizes, int n_in,
                              void* d_out, int out_size, void* d_ws, size_t ws_size,
                              hipStream_t stream) {
    (void)in_sizes; (void)n_in; (void)out_size; (void)d_ws; (void)ws_size;
    const float* y  = (const float*)d_in[0];
    const float* cb = (const float*)d_in[1];

    float* out = (float*)d_out;
    const size_t nElem = (size_t)BB * CC * HWSZ;   // 6,291,456
    float* out_uc  = out;
    float* out_yba = out + nElem;
    float* out_ci  = out + 2 * nElem;              // B*1*G*H*W = 262,144 floats

    const int blocks = BB * GG * (HWSZ / ROWS_PER_BLOCK);   // 8*8*32 = 2048
    vq_nearest_wmma_kernel<<<dim3(blocks), dim3(256), 0, stream>>>(
        y, cb, out_uc, out_yba, out_ci);
}
